// SelfAttention_20710332301552
// MI455X (gfx1250) — matrix operand order repaired and checked by reading, not compiled, not run on hardware
//
#include <hip/hip_runtime.h>
#include <math.h>

// ---------------------------------------------------------------------------
// Self-attention (B=8, S=2048, D=1024) for gfx1250 (wave32, bf16 WMMA).
//
// Pipeline:
//   0) cvt_f32_to_bf16:  x -> xb, Wq/Wk/Wv -> bf16   (one-shot)
//   1) qkv_gemm:   Q/K/V = xb @ Wb^T + b    (bf16 WMMA, fp32 accum)
//   2) per batch (stream-serialized, buffers reused):
//      scores_gemm:       S = (Q K^T) * scale           (bf16 WMMA -> fp32)
//      col_softmax_stats: per-column (query-axis) max & 1/sum(exp)
//      softmax_normalize: P = exp(S - m[k]) * rz[k] -> bf16 (coalesced pass)
//      transpose_bf16:    Vt = V^T (LDS-tiled)  -> contiguous B fragments
//      pv_gemm:           out = P @ V            (pure bf16 WMMA GEMM)
//
// All GEMMs: 32x64 wave tiles (2x4 register blocking), 8 WMMAs per 6 b128
// fragment loads, ~130 VGPRs/wave (no scratch spills).
//
// Workspace (aliased): Q/K/V bf16 3x33.5MB
//   | xregion 33.5MB = {xb} then {Sb 16.8 + Pbf 8.4 + Vt 4.2 + stats}
//   | W bf16 3x2MB   ->  ~140.5 MB total.
// ---------------------------------------------------------------------------

typedef __bf16 bf16_t;
typedef __attribute__((ext_vector_type(16))) __bf16 v16bf;
typedef __attribute__((ext_vector_type(8)))  __bf16 v8bf;
typedef __attribute__((ext_vector_type(8)))  float  v8f;

#define B_ 8
#define S_ 2048
#define D_ 1024

// ---- WMMA fragment loaders (wave32 layouts per CDNA5 ISA 7.12.2) ----------
// A (16x32 bf16): half = lane>>4, m = lane&15;
//   elems 0..15 -> k = kb + half*16 + j (contiguous, matches B)
// B (32x16 bf16): n = lane&15; elems 0..15 -> k = kb + half*16 + j (contiguous)

__device__ __forceinline__ v16bf load_a_bf16(const bf16_t* __restrict__ base, int ld,
                                             int row0, int kb, int lane) {
  const int half = lane >> 4, r = lane & 15;
  const bf16_t* p = base + (size_t)(row0 + r) * ld + kb + half * 16;
  v8bf lo = *(const v8bf*)p;          // k offsets 0..7   (16B load)
  v8bf hi = *(const v8bf*)(p + 8);    // k offsets 8..15  (16B load)
  return __builtin_shufflevector(lo, hi, 0, 1, 2, 3, 4, 5, 6, 7,
                                 8, 9, 10, 11, 12, 13, 14, 15);
}

__device__ __forceinline__ v16bf load_b_bf16(const bf16_t* __restrict__ base, int ld,
                                             int col0, int kb, int lane) {
  const int half = lane >> 4, n = lane & 15;
  const bf16_t* p = base + (size_t)(col0 + n) * ld + kb + half * 16;
  v8bf lo = *(const v8bf*)p;          // k 0..7  (16B load)
  v8bf hi = *(const v8bf*)(p + 8);    // k 8..15 (16B load)
  return __builtin_shufflevector(lo, hi, 0, 1, 2, 3, 4, 5, 6, 7,
                                 8, 9, 10, 11, 12, 13, 14, 15);
}

// ---- Kernel 0: one-shot fp32 -> bf16 --------------------------------------
__global__ void cvt_f32_to_bf16(const float* __restrict__ in,
                                bf16_t* __restrict__ out, int n) {
  const int i = (blockIdx.x * blockDim.x + threadIdx.x) * 8;
  if (i >= n) return;
  v8bf o;
#pragma unroll
  for (int j = 0; j < 8; ++j) o[j] = (bf16_t)in[i + j];
  *(v8bf*)(out + i) = o;
}

// ---- Kernel 1: QKV projection (32x64 wave tile, 2x4 register blocking) ----
// grid = (D/256, B*S/32, 3), block = 128 (4 waves, each a 32x64 tile)
__global__ void __launch_bounds__(128)
qkv_gemm(const bf16_t* __restrict__ xb,
         const bf16_t* __restrict__ Wqb, const float* __restrict__ bq,
         const bf16_t* __restrict__ Wkb, const float* __restrict__ bk,
         const bf16_t* __restrict__ Wvb, const float* __restrict__ bv,
         bf16_t* __restrict__ Q, bf16_t* __restrict__ K,
         bf16_t* __restrict__ V) {
  const int lane = threadIdx.x & 31;
  const int wave = threadIdx.x >> 5;
  const int m0 = blockIdx.y * 32;
  const int n0 = (blockIdx.x * 4 + wave) * 64;

  const bf16_t* W;  const float* bias;  bf16_t* outp;
  if (blockIdx.z == 0)      { W = Wqb; bias = bq; outp = Q; }
  else if (blockIdx.z == 1) { W = Wkb; bias = bk; outp = K; }
  else                      { W = Wvb; bias = bv; outp = V; }

  v8f acc[2][4] = {};
  for (int kb = 0; kb < D_; kb += 32) {
    v16bf a0 = load_a_bf16(xb, D_, m0, kb, lane);
    v16bf a1 = load_a_bf16(xb, D_, m0 + 16, kb, lane);
#pragma unroll
    for (int j = 0; j < 4; ++j) {
      v16bf bj = load_b_bf16(W, D_, n0 + 16 * j, kb, lane);
      acc[0][j] = __builtin_amdgcn_wmma_f32_16x16x32_bf16(
          false, a0, false, bj, (short)0, acc[0][j], false, false);
      acc[1][j] = __builtin_amdgcn_wmma_f32_16x16x32_bf16(
          false, a1, false, bj, (short)0, acc[1][j], false, false);
    }
  }

  const int half = lane >> 4, c = lane & 15;
#pragma unroll
  for (int j = 0; j < 4; ++j) {
    const int n = n0 + 16 * j + c;
    const float bn = bias[n];
#pragma unroll
    for (int i = 0; i < 2; ++i) {
#pragma unroll
      for (int r = 0; r < 8; ++r) {
        const int m = m0 + 16 * i + half * 8 + r;
        outp[(size_t)m * D_ + n] = (bf16_t)(acc[i][j][r] + bn);
      }
    }
  }
}

// ---- Kernel 2: scores = Q K^T * scale (one batch, 2x4 blocked) ------------
// grid = (S/256, S/32), block = 128
__global__ void __launch_bounds__(128)
scores_gemm(const bf16_t* __restrict__ Qb,
            const bf16_t* __restrict__ Kb,
            float* __restrict__ Sb, float scale) {
  const int lane = threadIdx.x & 31;
  const int wave = threadIdx.x >> 5;
  const int q0 = blockIdx.y * 32;
  const int k0 = (blockIdx.x * 4 + wave) * 64;

  v8f acc[2][4] = {};
  for (int kb = 0; kb < D_; kb += 32) {
    v16bf a0 = load_a_bf16(Qb, D_, q0, kb, lane);
    v16bf a1 = load_a_bf16(Qb, D_, q0 + 16, kb, lane);
#pragma unroll
    for (int j = 0; j < 4; ++j) {
      v16bf bj = load_b_bf16(Kb, D_, k0 + 16 * j, kb, lane);
      acc[0][j] = __builtin_amdgcn_wmma_f32_16x16x32_bf16(
          false, a0, false, bj, (short)0, acc[0][j], false, false);
      acc[1][j] = __builtin_amdgcn_wmma_f32_16x16x32_bf16(
          false, a1, false, bj, (short)0, acc[1][j], false, false);
    }
  }

  const int half = lane >> 4, c = lane & 15;
#pragma unroll
  for (int i = 0; i < 2; ++i)
#pragma unroll
    for (int j = 0; j < 4; ++j)
#pragma unroll
      for (int r = 0; r < 8; ++r) {
        const int q = q0 + 16 * i + half * 8 + r;
        Sb[(size_t)q * S_ + k0 + 16 * j + c] = acc[i][j][r] * scale;
      }
}

// ---- Kernel 3: per-column (query-axis) softmax stats ----------------------
// Reference softmaxes over axis=1 (queries): normalize each k-column.
// grid = S/64 blocks, 256 threads (64 columns x 4 row-strides).
__global__ void col_softmax_stats(const float* __restrict__ Sb,
                                  float* __restrict__ cmax,
                                  float* __restrict__ crz) {
  __shared__ float red[4][64];
  __shared__ float cmS[64];
  const int tx = threadIdx.x & 63;
  const int ty = threadIdx.x >> 6;
  const int k = blockIdx.x * 64 + tx;

  float m = -3.4e38f;
  for (int q = ty; q < S_; q += 4) m = fmaxf(m, Sb[(size_t)q * S_ + k]);
  red[ty][tx] = m;
  __syncthreads();
  if (ty == 0) {
    float mm = fmaxf(fmaxf(red[0][tx], red[1][tx]),
                     fmaxf(red[2][tx], red[3][tx]));
    cmS[tx] = mm;
    cmax[k] = mm;
  }
  __syncthreads();
  const float mm = cmS[tx];
  float s = 0.f;
  for (int q = ty; q < S_; q += 4) s += __expf(Sb[(size_t)q * S_ + k] - mm);
  __syncthreads();
  red[ty][tx] = s;
  __syncthreads();
  if (ty == 0) {
    float ss = red[0][tx] + red[1][tx] + red[2][tx] + red[3][tx];
    crz[k] = 1.0f / ss;
  }
}

// ---- Kernel 3b: P = exp(S - m[k]) * rz[k] -> bf16 (coalesced) -------------
// grid = S*S/(8*256) blocks, 256 threads, 8 elems/thread
__global__ void softmax_normalize(const float* __restrict__ Sb,
                                  const float* __restrict__ cmax,
                                  const float* __restrict__ crz,
                                  bf16_t* __restrict__ P) {
  const size_t idx = ((size_t)blockIdx.x * blockDim.x + threadIdx.x) * 8;
  const int k = (int)(idx & (S_ - 1));
  const float* s = Sb + idx;
  v8bf o;
#pragma unroll
  for (int j = 0; j < 8; ++j)
    o[j] = (bf16_t)(__expf(s[j] - cmax[k + j]) * crz[k + j]);
  *(v8bf*)(P + idx) = o;
}

// ---- Kernel 3c: Vt = V^T (bf16, LDS-tiled 32x32) --------------------------
// grid = (S/32, D/32), block = (32, 8)
__global__ void transpose_bf16(const bf16_t* __restrict__ in,
                               bf16_t* __restrict__ out) {
  __shared__ bf16_t tile[32][33];
  const int k0 = blockIdx.x * 32;   // row block in V (token index)
  const int d0 = blockIdx.y * 32;   // col block in V (feature index)
  const int tx = threadIdx.x;
  const int ty = threadIdx.y;
#pragma unroll
  for (int i = 0; i < 4; ++i)
    tile[ty + 8 * i][tx] = in[(size_t)(k0 + ty + 8 * i) * D_ + d0 + tx];
  __syncthreads();
#pragma unroll
  for (int i = 0; i < 4; ++i)
    out[(size_t)(d0 + ty + 8 * i) * S_ + k0 + tx] = tile[tx][ty + 8 * i];
}

// ---- Kernel 4: out = P @ V (one batch, pure bf16 GEMM, 2x4 blocked) -------
// grid = (D/256, S/32), block = 128. A = P (q,k), B = Vt (d,k): all loads b128.
__global__ void __launch_bounds__(128)
pv_gemm(const bf16_t* __restrict__ Pb,
        const bf16_t* __restrict__ Vt,
        float* __restrict__ Out) {
  const int lane = threadIdx.x & 31;
  const int wave = threadIdx.x >> 5;
  const int q0 = blockIdx.y * 32;
  const int d0 = (blockIdx.x * 4 + wave) * 64;

  v8f acc[2][4] = {};
  for (int kb = 0; kb < S_; kb += 32) {
    v16bf a0 = load_a_bf16(Pb, S_, q0, kb, lane);
    v16bf a1 = load_a_bf16(Pb, S_, q0 + 16, kb, lane);
#pragma unroll
    for (int j = 0; j < 4; ++j) {
      v16bf bj = load_b_bf16(Vt, S_, d0 + 16 * j, kb, lane);
      acc[0][j] = __builtin_amdgcn_wmma_f32_16x16x32_bf16(
          false, a0, false, bj, (short)0, acc[0][j], false, false);
      acc[1][j] = __builtin_amdgcn_wmma_f32_16x16x32_bf16(
          false, a1, false, bj, (short)0, acc[1][j], false, false);
    }
  }

  const int half = lane >> 4, c = lane & 15;
#pragma unroll
  for (int i = 0; i < 2; ++i)
#pragma unroll
    for (int j = 0; j < 4; ++j)
#pragma unroll
      for (int r = 0; r < 8; ++r) {
        const int q = q0 + 16 * i + half * 8 + r;
        Out[(size_t)q * D_ + d0 + 16 * j + c] = acc[i][j][r];
      }
}

// ---------------------------------------------------------------------------
extern "C" void kernel_launch(void* const* d_in, const int* in_sizes, int n_in,
                              void* d_out, int out_size, void* d_ws, size_t ws_size,
                              hipStream_t stream) {
  const float* x  = (const float*)d_in[0];
  const float* Wq = (const float*)d_in[1];
  const float* bq = (const float*)d_in[2];
  const float* Wk = (const float*)d_in[3];
  const float* bk = (const float*)d_in[4];
  const float* Wv = (const float*)d_in[5];
  const float* bv = (const float*)d_in[6];
  float* out = (float*)d_out;

  char* ws = (char*)d_ws;
  const size_t qkvBytes = (size_t)B_ * S_ * D_ * sizeof(bf16_t);   // 33.5 MB
  bf16_t* Qb = (bf16_t*)(ws);
  bf16_t* Kb = (bf16_t*)(ws + qkvBytes);
  bf16_t* Vb = (bf16_t*)(ws + 2 * qkvBytes);
  // Aliased region: xb (bf16 copy of x) is dead after qkv_gemm; the per-batch
  // scratch (Sb 16.8MB + Pbf 8.4MB + Vt 4.2MB + stats) reuses its 33.5MB.
  char* xregion = ws + 3 * qkvBytes;
  bf16_t* xb   = (bf16_t*)xregion;
  const size_t sBytes  = (size_t)S_ * S_ * sizeof(float);    // 16.8 MB
  const size_t pBytes  = (size_t)S_ * S_ * sizeof(bf16_t);   //  8.4 MB
  const size_t vtBytes = (size_t)S_ * D_ * sizeof(bf16_t);   //  4.2 MB
  float*  Sb   = (float*)xregion;
  bf16_t* Pbf  = (bf16_t*)(xregion + sBytes);
  bf16_t* Vt   = (bf16_t*)(xregion + sBytes + pBytes);
  float*  cmax = (float*)(xregion + sBytes + pBytes + vtBytes);
  float*  crz  = cmax + S_;
  bf16_t* Wqb = (bf16_t*)(ws + 4 * qkvBytes);
  bf16_t* Wkb = Wqb + (size_t)D_ * D_;
  bf16_t* Wvb = Wkb + (size_t)D_ * D_;

  // 0) One-shot conversions to bf16
  {
    const int nx = B_ * S_ * D_;
    const int nw = D_ * D_;
    cvt_f32_to_bf16<<<nx / 2048, 256, 0, stream>>>(x, xb, nx);
    cvt_f32_to_bf16<<<nw / 2048, 256, 0, stream>>>(Wq, Wqb, nw);
    cvt_f32_to_bf16<<<nw / 2048, 256, 0, stream>>>(Wk, Wkb, nw);
    cvt_f32_to_bf16<<<nw / 2048, 256, 0, stream>>>(Wv, Wvb, nw);
  }

  const dim3 blk(128);

  // 1) QKV projections (all batches, all three weights)
  qkv_gemm<<<dim3(D_ / 256, (B_ * S_) / 32, 3), blk, 0, stream>>>(
      xb, Wqb, bq, Wkb, bk, Wvb, bv, Qb, Kb, Vb);

  const float scale = 1.0f / (sqrtf((float)D_) + 1e-6f);

  // 2) Per-batch attention (serialized on the stream; buffers reused)
  for (int b = 0; b < B_; ++b) {
    const bf16_t* Qbb = Qb + (size_t)b * S_ * D_;
    const bf16_t* Kbb = Kb + (size_t)b * S_ * D_;
    const bf16_t* Vbb = Vb + (size_t)b * S_ * D_;
    float* outb = out + (size_t)b * S_ * D_;

    scores_gemm<<<dim3(S_ / 256, S_ / 32), blk, 0, stream>>>(Qbb, Kbb, Sb, scale);
    col_softmax_stats<<<dim3(S_ / 64), dim3(256), 0, stream>>>(Sb, cmax, crz);
    softmax_normalize<<<(S_ * S_) / 2048, 256, 0, stream>>>(Sb, cmax, crz, Pbf);
    transpose_bf16<<<dim3(S_ / 32, D_ / 32), dim3(32, 8), 0, stream>>>(Vbb, Vt);
    pv_gemm<<<dim3(D_ / 256, S_ / 32), blk, 0, stream>>>(Pbf, Vt, outb);
  }
}
